// HamiltonianBlock_38336878084588
// MI455X (gfx1250) — matrix operand order repaired and checked by reading, not compiled, not run on hardware
//
#include <hip/hip_runtime.h>
#include <hip/hip_bf16.h>

typedef __attribute__((ext_vector_type(16))) _Float16 v16h;
typedef __attribute__((ext_vector_type(8)))  _Float16 v8h;
typedef __attribute__((ext_vector_type(8)))  float    v8f;

#define B_    8
#define S_    4096
#define D_    1024
#define HID_  4096
#define K2_   2048          // 2*D
#define TM_   64            // tokens per workgroup (4 m-tiles of 16)
#define LDA_  2056          // padded LDS row stride in halves (2048 + 8)
#define NBLK_ ((B_ * S_) / TM_)   // 512 workgroups for the MLP GEMM

// CDNA5 has V_TANH_F32 (trans op); use it if the toolchain exposes it.
__device__ __forceinline__ float fast_tanh(float x) {
#if __has_builtin(__builtin_amdgcn_tanhf)
    return __builtin_amdgcn_tanhf(x);
#elif __has_builtin(__builtin_amdgcn_tanh_f32)
    return __builtin_amdgcn_tanh_f32(x);
#else
    return tanhf(x);
#endif
}

// A fragment (contiguous K, matching B): halves[0..15] = K kb+16*hi..+15.
// `off` is a compile-time-foldable
// half-offset (row select + kb) from the lane's LDS base pointer.
__device__ __forceinline__ v16h ld_frag_a(const _Float16* __restrict__ base,
                                          int off, int hi) {
    v8h lo = *(const v8h*)(base + off + 16 * hi);
    v8h hh = *(const v8h*)(base + off + 16 * hi + 8);
    return __builtin_shufflevector(lo, hh, 0, 1, 2, 3, 4, 5, 6, 7,
                                   8, 9, 10, 11, 12, 13, 14, 15);
}
// B fragment: lane = column n, 16 contiguous K starting kb (+16*hi folded
// into the row pointer).
__device__ __forceinline__ v16h ld_frag_b(const _Float16* __restrict__ brow,
                                          int kb) {
    v8h lo = *(const v8h*)(brow + kb);
    v8h hh = *(const v8h*)(brow + kb + 8);
    return __builtin_shufflevector(lo, hh, 0, 1, 2, 3, 4, 5, 6, 7,
                                   8, 9, 10, 11, 12, 13, 14, 15);
}

#define WMMA16(A, Bf, C) __builtin_amdgcn_wmma_f32_16x16x32_f16( \
        false, (A), false, (Bf), (short)0, (C), false, false)

// LDS row permutation: token t -> row (t&15)*4 + (t>>4), so a lane's four
// m-tile rows (lm, lm+16, lm+32, lm+48) are consecutive rows at lm*4.
__device__ __forceinline__ int lds_row(int t) { return (t & 15) * 4 + (t >> 4); }

// ---------------------------------------------------------------------------
// w1 [HID, 2D] f32 -> f16 (stays L2-resident afterwards: 16.8 MB)
// ---------------------------------------------------------------------------
__global__ __launch_bounds__(256) void k_conv_w1(const float* __restrict__ w1,
                                                 _Float16* __restrict__ w1h, int n) {
    int i = blockIdx.x * 256 + threadIdx.x;
    if (i < n) w1h[i] = (_Float16)w1[i];
}

// ---------------------------------------------------------------------------
// qs/ps partial token sums: grid (32 index-chunks, 16 s-chunks), deterministic
// ---------------------------------------------------------------------------
__global__ __launch_bounds__(256) void k_sums_part(const float* __restrict__ q,
                                                   const float* __restrict__ p,
                                                   float* __restrict__ qs_part,
                                                   float* __restrict__ ps_part) {
    int i  = blockIdx.x * 256 + threadIdx.x;   // 0..B*D-1 (8192)
    int sc = blockIdx.y;                       // 0..15
    int b = i >> 10;
    int d = i & (D_ - 1);
    const float* qp = q + ((long)(b * S_ + sc * 256)) * D_ + d;
    const float* pp = p + ((long)(b * S_ + sc * 256)) * D_ + d;
    float sq = 0.f, sp = 0.f;
    for (int s = 0; s < 256; ++s) {
        sq += qp[(long)s * D_];
        sp += pp[(long)s * D_];
    }
    qs_part[sc * (B_ * D_) + i] = sq;
    ps_part[sc * (B_ * D_) + i] = sp;
}

__global__ __launch_bounds__(256) void k_sums_fin(const float* __restrict__ qs_part,
                                                  const float* __restrict__ ps_part,
                                                  float* __restrict__ qs,
                                                  float* __restrict__ ps) {
    int i = blockIdx.x * 256 + threadIdx.x;    // 0..8191
    float sq = 0.f, sp = 0.f;
    for (int sc = 0; sc < 16; ++sc) {
        sq += qs_part[sc * (B_ * D_) + i];
        sp += ps_part[sc * (B_ * D_) + i];
    }
    qs[i] = sq;
    ps[i] = sp;
}

// ---------------------------------------------------------------------------
// Row sums of cqp/cqq/cpp (sum over axis=1): one block per (matrix, row)
// ---------------------------------------------------------------------------
__global__ __launch_bounds__(256) void k_rowsum(const float* __restrict__ cqp,
                                                const float* __restrict__ cqq,
                                                const float* __restrict__ cpp,
                                                float* __restrict__ rs) {
    __shared__ float red[256];
    int bx  = blockIdx.x;
    int mat = bx >> 10;
    int d   = bx & (D_ - 1);
    const float* src = (mat == 0) ? cqp : (mat == 1) ? cqq : cpp;
    float a = 0.f;
    for (int k = threadIdx.x; k < D_; k += 256) a += src[(long)d * D_ + k];
    red[threadIdx.x] = a;
    __syncthreads();
    for (int st = 128; st > 0; st >>= 1) {
        if (threadIdx.x < st) red[threadIdx.x] += red[threadIdx.x + st];
        __syncthreads();
    }
    if (threadIdx.x == 0) rs[mat * D_ + d] = red[0];
}

// ---------------------------------------------------------------------------
// MLP GEMM+tanh+reduce: M=32768 tokens, K=2048, N=4096 via WMMA f16->f32.
// 8 waves/WG. Wave w register-blocks 4 m-tiles x 2 n-tiles (8 accumulators):
// per k-step, 4 A + 2 B fragments feed 8 WMMAs. Each WG reads w1h exactly
// once (L2-resident). A tile (64 tokens x full K, f16) lives in 263 KB LDS
// (CDNA5 320KB/WGP). Inner loop is software-pipelined with ping-pong
// fragment buffers (unroll-2) so rotation is pure register renaming.
// ---------------------------------------------------------------------------
__global__ __launch_bounds__(256) void k_mlp(const float* __restrict__ q,
                                             const float* __restrict__ p,
                                             const _Float16* __restrict__ w1h,
                                             const float* __restrict__ b1,
                                             const float* __restrict__ w2,
                                             float* __restrict__ mlp_part) {
    extern __shared__ _Float16 Ash[];          // [TM_][LDA_], permuted rows
    const int tid = threadIdx.x;
    const int m0  = blockIdx.x * TM_;
    const int b   = m0 / S_;                   // whole WG shares one batch
    const int s0  = m0 % S_;

    // ---- stage A = concat(q,p) rows as f16 into LDS (coalesced over d) ----
    for (int idx = tid; idx < TM_ * D_; idx += 256) {
        int t = idx >> 10;
        int d = idx & (D_ - 1);
        long g = ((long)(b * S_ + s0 + t)) * D_ + d;
        int r = lds_row(t);
        Ash[r * LDA_ + d]      = (_Float16)q[g];
        Ash[r * LDA_ + D_ + d] = (_Float16)p[g];
    }
    __syncthreads();

    const int w  = tid >> 5;
    const int l  = tid & 31;
    const int hi = l >> 4;
    const int lm = l & 15;

    // one LDS base per lane; its 4 m-tile rows are at +0,+LDA_,+2*LDA_,+3*LDA_
    const _Float16* abase = &Ash[(lm * 4) * LDA_];

    float acc = 0.f;

    for (int j = 0; j < HID_ / 256; ++j) {     // 16 iterations
        const int n0 = j * 256 + w * 32 + lm;  // this lane's two columns
        const int n1 = n0 + 16;
        const _Float16* brow0 = w1h + (long)n0 * K2_ + 16 * hi;
        const _Float16* brow1 = w1h + (long)n1 * K2_ + 16 * hi;

        v8f c00 = {}, c01 = {}, c02 = {}, c03 = {};
        v8f c10 = {}, c11 = {}, c12 = {}, c13 = {};

        // prologue: buffer X <- kb = 0
        v16h xA0 = ld_frag_a(abase, 0 * LDA_, hi);
        v16h xA1 = ld_frag_a(abase, 1 * LDA_, hi);
        v16h xA2 = ld_frag_a(abase, 2 * LDA_, hi);
        v16h xA3 = ld_frag_a(abase, 3 * LDA_, hi);
        v16h xB0 = ld_frag_b(brow0, 0);
        v16h xB1 = ld_frag_b(brow1, 0);

        #pragma unroll 1
        for (int kb = 0; kb < K2_; kb += 64) {
            // fill buffer Y from kb+32 (independent of X's consumers)
            v16h yA0 = ld_frag_a(abase, 0 * LDA_ + kb + 32, hi);
            v16h yA1 = ld_frag_a(abase, 1 * LDA_ + kb + 32, hi);
            v16h yA2 = ld_frag_a(abase, 2 * LDA_ + kb + 32, hi);
            v16h yA3 = ld_frag_a(abase, 3 * LDA_ + kb + 32, hi);
            v16h yB0 = ld_frag_b(brow0, kb + 32);
            v16h yB1 = ld_frag_b(brow1, kb + 32);
            __builtin_prefetch(brow0 + kb + 96, 0, 3);
            __builtin_prefetch(brow1 + kb + 96, 0, 3);

            // compute on X (loaded a half-iteration ago)
            c00 = WMMA16(xA0, xB0, c00);
            c01 = WMMA16(xA1, xB0, c01);
            c02 = WMMA16(xA2, xB0, c02);
            c03 = WMMA16(xA3, xB0, c03);
            c10 = WMMA16(xA0, xB1, c10);
            c11 = WMMA16(xA1, xB1, c11);
            c12 = WMMA16(xA2, xB1, c12);
            c13 = WMMA16(xA3, xB1, c13);

            // refill X from kb+64 (one-past-end reads on last iter are benign)
            xA0 = ld_frag_a(abase, 0 * LDA_ + kb + 64, hi);
            xA1 = ld_frag_a(abase, 1 * LDA_ + kb + 64, hi);
            xA2 = ld_frag_a(abase, 2 * LDA_ + kb + 64, hi);
            xA3 = ld_frag_a(abase, 3 * LDA_ + kb + 64, hi);
            xB0 = ld_frag_b(brow0, kb + 64);
            xB1 = ld_frag_b(brow1, kb + 64);

            // compute on Y
            c00 = WMMA16(yA0, yB0, c00);
            c01 = WMMA16(yA1, yB0, c01);
            c02 = WMMA16(yA2, yB0, c02);
            c03 = WMMA16(yA3, yB0, c03);
            c10 = WMMA16(yA0, yB1, c10);
            c11 = WMMA16(yA1, yB1, c11);
            c12 = WMMA16(yA2, yB1, c12);
            c13 = WMMA16(yA3, yB1, c13);
        }

        // epilogue: +b1, v_tanh_f32, dot with w2 column weights
        const float bn0 = b1[n0], bn1 = b1[n1];
        const float w20 = w2[n0], w21 = w2[n1];
        float s0 = 0.f, s1 = 0.f;
        #pragma unroll
        for (int r = 0; r < 8; ++r) {
            s0 += fast_tanh(c00[r] + bn0);
            s0 += fast_tanh(c01[r] + bn0);
            s0 += fast_tanh(c02[r] + bn0);
            s0 += fast_tanh(c03[r] + bn0);
            s1 += fast_tanh(c10[r] + bn1);
            s1 += fast_tanh(c11[r] + bn1);
            s1 += fast_tanh(c12[r] + bn1);
            s1 += fast_tanh(c13[r] + bn1);
        }
        acc += s0 * w20 + s1 * w21;
    }

    // wave32 reduction, then block reduction -> per-block partial (no atomics)
    for (int off = 16; off > 0; off >>= 1) acc += __shfl_xor(acc, off, 32);
    __syncthreads();
    float* red = (float*)Ash;                  // reuse LDS
    if (l == 0) red[w] = acc;
    __syncthreads();
    if (tid == 0) {
        float t = 0.f;
        for (int i = 0; i < 8; ++i) t += red[i];
        mlp_part[blockIdx.x] = t;
    }
}

// ---------------------------------------------------------------------------
// Final scalar: B*h_offset + B*S*b2 + lin + quad + sum(mlp partials)
// ---------------------------------------------------------------------------
__global__ __launch_bounds__(256) void k_final(const float* __restrict__ qs,
                                               const float* __restrict__ ps,
                                               const float* __restrict__ rs,
                                               const float* __restrict__ clq,
                                               const float* __restrict__ clp,
                                               const float* __restrict__ mlp_part,
                                               const float* __restrict__ h_offset,
                                               const float* __restrict__ b2,
                                               float* __restrict__ out_scalar) {
    __shared__ float red[256];
    int tid = threadIdx.x;
    float a = 0.f;
    for (int i = tid; i < B_ * D_; i += 256) {
        int d = i & (D_ - 1);
        float qv = qs[i], pv = ps[i];
        a += qv * clq[d] + pv * clp[d];                                 // lin
        a += qv * rs[d] * pv + qv * rs[D_ + d] * qv + pv * rs[2 * D_ + d] * pv;  // quad
    }
    for (int i = tid; i < NBLK_; i += 256) a += mlp_part[i];
    red[tid] = a;
    __syncthreads();
    for (int st = 128; st > 0; st >>= 1) {
        if (tid < st) red[tid] += red[tid + st];
        __syncthreads();
    }
    if (tid == 0)
        out_scalar[0] = red[0] + (float)B_ * h_offset[0]
                               + (float)B_ * (float)S_ * b2[0];
}

// ---------------------------------------------------------------------------
extern "C" void kernel_launch(void* const* d_in, const int* in_sizes, int n_in,
                              void* d_out, int out_size, void* d_ws, size_t ws_size,
                              hipStream_t stream) {
    const float* q    = (const float*)d_in[0];
    const float* p    = (const float*)d_in[1];
    const float* w1   = (const float*)d_in[2];
    const float* b1   = (const float*)d_in[3];
    const float* w2   = (const float*)d_in[4];
    const float* b2   = (const float*)d_in[5];
    const float* hoff = (const float*)d_in[6];
    const float* clq  = (const float*)d_in[7];
    const float* clp  = (const float*)d_in[8];
    const float* cqp  = (const float*)d_in[9];
    const float* cqq  = (const float*)d_in[10];
    const float* cpp  = (const float*)d_in[11];
    float* out = (float*)d_out;

    const size_t BSD = (size_t)B_ * S_ * D_;

    // pass-through outputs (q, p) — async D2D, graph-capture safe
    hipMemcpyAsync(out,        q, BSD * sizeof(float), hipMemcpyDeviceToDevice, stream);
    hipMemcpyAsync(out + BSD,  p, BSD * sizeof(float), hipMemcpyDeviceToDevice, stream);

    // workspace layout (bytes)
    char* ws = (char*)d_ws;
    size_t off = 0;
    _Float16* w1h     = (_Float16*)(ws + off); off += (size_t)HID_ * K2_ * sizeof(_Float16); // 16.8 MB
    float*    qs_part = (float*)(ws + off);    off += (size_t)16 * B_ * D_ * sizeof(float);
    float*    ps_part = (float*)(ws + off);    off += (size_t)16 * B_ * D_ * sizeof(float);
    float*    qs      = (float*)(ws + off);    off += (size_t)B_ * D_ * sizeof(float);
    float*    ps      = (float*)(ws + off);    off += (size_t)B_ * D_ * sizeof(float);
    float*    rs      = (float*)(ws + off);    off += (size_t)3 * D_ * sizeof(float);
    float*    mlp_p   = (float*)(ws + off);    off += (size_t)NBLK_ * sizeof(float);

    const int w1n = HID_ * K2_;
    k_conv_w1<<<(w1n + 255) / 256, 256, 0, stream>>>(w1, w1h, w1n);
    k_sums_part<<<dim3((B_ * D_) / 256, 16), 256, 0, stream>>>(q, p, qs_part, ps_part);
    k_sums_fin<<<(B_ * D_) / 256, 256, 0, stream>>>(qs_part, ps_part, qs, ps);
    k_rowsum<<<3 * D_, 256, 0, stream>>>(cqp, cqq, cpp, rs);

    const size_t lds_bytes = (size_t)TM_ * LDA_ * sizeof(_Float16);  // 263,168 B (<320KB WGP)
    (void)hipFuncSetAttribute((const void*)k_mlp,
                              hipFuncAttributeMaxDynamicSharedMemorySize,
                              (int)lds_bytes);
    k_mlp<<<NBLK_, 256, lds_bytes, stream>>>(q, p, w1h, b1, w2, mlp_p);

    k_final<<<1, 256, 0, stream>>>(qs, ps, rs, clq, clp, mlp_p, hoff, b2, out + 2 * BSD);
}
